// MSA2_69114613728827
// MI455X (gfx1250) — matrix operand order repaired and checked by reading, not compiled, not run on hardware
//
#include <hip/hip_runtime.h>
#include <hip/hip_bf16.h>
#include <math.h>

// ---------------------------------------------------------------------------
// Single-query multi-head cross attention, restructured:
//   qk[b,n,c]  = (1/8) * WK[:,n*64:(n+1)*64] @ (x[b]@WQ + WQb)[n]      (tiny)
//   scores^T   = y @ qk^T            (4.3 GFLOP, WMMA bf16)
//   online softmax over L (flash-style, 16 L-slices combined later)
//   ybar^T     += y^T @ alpha^T      (4.3 GFLOP, WMMA bf16, K never materialized)
//   att        = (ybar/S) @ WV + WVb ; out = att @ WO + WOb
// WK_b is softmax-invariant (uniform shift over L) and omitted.
// ---------------------------------------------------------------------------

typedef __attribute__((ext_vector_type(16))) __bf16 v16bf;
typedef __attribute__((ext_vector_type(8)))  __bf16 v8bf;
typedef __attribute__((ext_vector_type(4)))  __bf16 v4bf;
typedef __attribute__((ext_vector_type(8)))  float  v8f;

#define B_  32
#define L_  4096
#define C_  1024
#define NH  16
#define HD  64

#define LSLICE 256          // L rows per workgroup (16 slices per batch)
#define LTILE  32           // rows per staged tile (= WMMA K for ybar)
#define QKS    1032         // LDS row stride (bf16 elems) for qk  [16][1032]
#define YS     1032         // LDS row stride for y row-major tile [32][1032]
#define YTS    32           // LDS row stride for y transposed     [1024][32]

// LDS layout for k2 (bytes)
#define OFF_QK   0                       // 16*1032*2   = 33024
#define OFF_YR   33024                   // 32*1032*2   = 66048
#define OFF_YT   (33024 + 66048)         // 1024*32*2   = 65536
#define OFF_PART 164608                  // 8*256*4     = 8192
#define OFF_SC   172800                  // 32*16*4     = 2048
#define OFF_ALP  174848                  // 16*32*2     = 1024
#define OFF_M    175872                  // 16*4
#define OFF_S    175936                  // 16*4
#define OFF_RS   176000                  // 16*4
#define SMEM2    176064

#define SMEM3    (1024 + 64 + C_*NH*4)   // 66624

// workspace layout (bytes)
#define QK_OFF   0u                                   // 32*16*1024*2 = 1 MiB
#define YBAR_OFF 1048576u                             // 512*16384*4  = 32 MiB
#define MS_OFF   (YBAR_OFF + 512u*16384u*4u)          // 512*32*4
#define ATT_OFF  (MS_OFF + 65536u)                    // 32*1024*4

__device__ __forceinline__ v16bf ld_frag(const __bf16* p0, const __bf16* p1) {
    v8bf lo = *(const v8bf*)p0;
    v8bf hi = *(const v8bf*)p1;
    return __builtin_shufflevector(lo, hi, 0,1,2,3,4,5,6,7,8,9,10,11,12,13,14,15);
}

// ---------------------------------------------------------------------------
// k1: q = x@WQ + WQb (scaled by 1/sqrt(64)); qk[b,n,c] = sum_d q[n,d]*WK[c,n*64+d]
// ---------------------------------------------------------------------------
__global__ __launch_bounds__(256) void k1_qk(const float* __restrict__ x,
                                             const float* __restrict__ WQ,
                                             const float* __restrict__ WQb,
                                             const float* __restrict__ WK,
                                             __bf16* __restrict__ qk_ws) {
    __shared__ float xs[C_];
    __shared__ float qs[C_];
    const int b = blockIdx.x, t = threadIdx.x;
    for (int i = t; i < C_; i += 256) xs[i] = x[(size_t)b * C_ + i];
    __syncthreads();
    for (int col = t; col < C_; col += 256) {
        float acc = WQb[col];
        for (int c = 0; c < C_; ++c) acc += xs[c] * WQ[(size_t)c * C_ + col];
        qs[col] = acc * 0.125f;                      // fold 1/sqrt(HD)
    }
    __syncthreads();
    for (int v = t; v < NH * C_; v += 256) {         // v = n*1024 + c
        const int n = v >> 10, cc = v & 1023;
        const float* wk = WK + (size_t)cc * C_ + n * HD;
        float acc = 0.f;
        for (int d = 0; d < HD; ++d) acc += qs[n * HD + d] * wk[d];
        qk_ws[(size_t)b * (NH * C_) + v] = (__bf16)acc;
    }
}

// ---------------------------------------------------------------------------
// k2: fused scores + online softmax + ybar accumulation (per batch, per L-slice)
// ---------------------------------------------------------------------------
__global__ __launch_bounds__(256) void k2_attn(const float* __restrict__ y,
                                               const __bf16* __restrict__ qk_ws,
                                               float* __restrict__ ybar_ws,
                                               float* __restrict__ ms_ws) {
    extern __shared__ char smem[];
    __bf16* qk   = (__bf16*)(smem + OFF_QK);
    __bf16* yt_r = (__bf16*)(smem + OFF_YR);
    __bf16* yt_t = (__bf16*)(smem + OFF_YT);
    float*  part = (float*)(smem + OFF_PART);
    float*  sc   = (float*)(smem + OFF_SC);
    __bf16* alp  = (__bf16*)(smem + OFF_ALP);
    float*  mvec = (float*)(smem + OFF_M);
    float*  svec = (float*)(smem + OFF_S);
    float*  rsv  = (float*)(smem + OFF_RS);

    const int t = threadIdx.x, lane = t & 31, wv = t >> 5;
    const int b = blockIdx.x >> 4, slice = blockIdx.x & 15;
    const int l0base = slice * LSLICE;

    if (t < NH) { mvec[t] = -__builtin_inff(); svec[t] = 0.f; }
    {   // qk[b] -> LDS [16][1032] bf16
        const __bf16* src = qk_ws + (size_t)b * (NH * C_);
        for (int v = t; v < 2048; v += 256) {
            int n = v >> 7, cb = (v & 127) * 8;
            *(uint4*)&qk[n * QKS + cb] = *(const uint4*)&src[n * C_ + cb];
        }
    }
    v8f yb[8];                                        // ybar^T accum: this wave owns c in [wv*128, wv*128+128)
    for (int i = 0; i < 8; ++i) yb[i] = (v8f){0,0,0,0,0,0,0,0};
    __syncthreads();

    const int srow = t >> 3;                          // staging: row 0..31
    const int scol = (t & 7) * 128;                   // staging: col block
    const int fm   = lane & 15;                       // frag row/col index
    const int a_ko = (lane < 16) ? 0 : 8;             // A-frag K chunk base
    const int b_ko = (lane < 16) ? 0 : 8;             // B-frag K base

    for (int it = 0; it < LSLICE / LTILE; ++it) {
        const int l0 = l0base + it * LTILE;
        {   // ---- stage y tile f32 -> bf16, row-major + transposed copies ----
            const float* yr = y + ((size_t)b * L_ + l0 + srow) * C_ + scol;
            if (it + 1 < LSLICE / LTILE) __builtin_prefetch(yr + LTILE * C_, 0, 0);
            for (int j = 0; j < 32; ++j) {
                float4 f = *(const float4*)(yr + 4 * j);
                v4bf h; h[0]=(__bf16)f.x; h[1]=(__bf16)f.y; h[2]=(__bf16)f.z; h[3]=(__bf16)f.w;
                const int c = scol + 4 * j;
                *(v4bf*)&yt_r[srow * YS + c] = h;
                yt_t[(c + 0) * YTS + srow] = h[0];
                yt_t[(c + 1) * YTS + srow] = h[1];
                yt_t[(c + 2) * YTS + srow] = h[2];
                yt_t[(c + 3) * YTS + srow] = h[3];
            }
        }
        __syncthreads();
        {   // ---- scores^T[l,n] = y·qk^T : wave -> one l-tile, 256-wide K slice ----
            v8f sacc = (v8f){0,0,0,0,0,0,0,0};
            const int mt  = wv & 1;
            const int c0w = (wv >> 1) * 256;
            const int arow = mt * 16 + fm;
            for (int ks = 0; ks < 8; ++ks) {
                const int c0 = c0w + ks * 32;
                v16bf af = ld_frag(&yt_r[arow * YS + c0 + a_ko],
                                   &yt_r[arow * YS + c0 + a_ko + 16]);
                v16bf bf = ld_frag(&qk[fm * QKS + c0 + b_ko],
                                   &qk[fm * QKS + c0 + b_ko + 16]);
                sacc = __builtin_amdgcn_wmma_f32_16x16x32_bf16(
                           false, af, false, bf, (short)0, sacc, false, false);
            }
            const int mb = (lane < 16) ? 0 : 8;       // C layout: VGPR j -> M=j(+8)
            for (int j = 0; j < 8; ++j)
                part[wv * 256 + (mb + j) * 16 + fm] = sacc[j];
        }
        __syncthreads();
        {   // ---- reduce 4 K-slice partials per l-tile ----
            const int ll = t >> 4, n = t & 15;
            for (int tile = 0; tile < 2; ++tile) {
                float s = part[(tile + 0) * 256 + ll * 16 + n]
                        + part[(tile + 2) * 256 + ll * 16 + n]
                        + part[(tile + 4) * 256 + ll * 16 + n]
                        + part[(tile + 6) * 256 + ll * 16 + n];
                sc[(tile * 16 + ll) * 16 + n] = s;
            }
        }
        __syncthreads();
        if (t < NH) {   // ---- online softmax update, one thread per head ----
            const int n = t;
            float mo = mvec[n], mx = mo;
            for (int l = 0; l < 32; ++l) mx = fmaxf(mx, sc[l * 16 + n]);
            float rf = __expf(mo - mx);               // first iter: exp(-inf)=0
            float ss = 0.f;
            for (int l = 0; l < 32; ++l) {
                float e = __expf(sc[l * 16 + n] - mx);
                alp[n * 32 + l] = (__bf16)e;
                ss += e;
            }
            svec[n] = svec[n] * rf + ss;
            mvec[n] = mx;
            rsv[n]  = rf;
        }
        __syncthreads();
        {   // ---- rescale accumulators; ybar^T += yt^T x alpha^T (K=32) ----
            const float rf = rsv[fm];                 // N(=head) is the lane dim in C layout
            v16bf bf = ld_frag(&alp[fm * 32 + b_ko], &alp[fm * 32 + b_ko + 16]);
            for (int mt2 = 0; mt2 < 8; ++mt2) {
                for (int j = 0; j < 8; ++j) yb[mt2][j] *= rf;
                const int crow = wv * 128 + mt2 * 16 + fm;
                v16bf af = ld_frag(&yt_t[crow * YTS + a_ko],
                                   &yt_t[crow * YTS + a_ko + 16]);
                yb[mt2] = __builtin_amdgcn_wmma_f32_16x16x32_bf16(
                              false, af, false, bf, (short)0, yb[mt2], false, false);
            }
        }
        __syncthreads();
    }
    {   // ---- write per-slice partials: ybar^T [c][n], plus (m, s) ----
        const int mb = (lane < 16) ? 0 : 8;
        float* dst = ybar_ws + (size_t)blockIdx.x * (C_ * NH);
        for (int mt2 = 0; mt2 < 8; ++mt2)
            for (int j = 0; j < 8; ++j) {
                const int c = wv * 128 + mt2 * 16 + mb + j;
                dst[c * 16 + fm] = yb[mt2][j];
            }
        if (t < NH) {
            ms_ws[blockIdx.x * 32 + t]      = mvec[t];
            ms_ws[blockIdx.x * 32 + 16 + t] = svec[t];
        }
    }
}

// ---------------------------------------------------------------------------
// k3: combine 16 L-slice partials per batch; att = (ybar/S)@WV + WVb
// ---------------------------------------------------------------------------
__global__ __launch_bounds__(256) void k3_comb(const float* __restrict__ ybar_ws,
                                               const float* __restrict__ ms_ws,
                                               const float* __restrict__ WV,
                                               const float* __restrict__ WVb,
                                               float* __restrict__ att_ws) {
    extern __shared__ char smem[];
    float* wco  = (float*)smem;                      // [16 j][16 n]
    float* invS = (float*)(smem + 1024);
    float* yl   = (float*)(smem + 1088);             // [1024 c][16 n]
    const int b = blockIdx.x, t = threadIdx.x;
    if (t < NH) {
        const int n = t;
        float M = -__builtin_inff();
        for (int j = 0; j < 16; ++j) M = fmaxf(M, ms_ws[(b * 16 + j) * 32 + n]);
        float S = 0.f;
        for (int j = 0; j < 16; ++j) {
            float w = __expf(ms_ws[(b * 16 + j) * 32 + n] - M);
            wco[j * 16 + n] = w;
            S += w * ms_ws[(b * 16 + j) * 32 + 16 + n];
        }
        invS[n] = 1.f / S;
    }
    __syncthreads();
    for (int v = t; v < C_ * NH; v += 256) {         // v = c*16 + n
        const int n = v & 15;
        float a = 0.f;
        for (int j = 0; j < 16; ++j)
            a += wco[j * 16 + n] * ybar_ws[(size_t)(b * 16 + j) * (C_ * NH) + v];
        yl[v] = a;
    }
    __syncthreads();
    for (int k = 0; k < 4; ++k) {
        const int nd = t + k * 256, n = nd >> 6;
        float acc = 0.f;
        for (int c = 0; c < C_; ++c) acc += yl[c * 16 + n] * WV[(size_t)c * C_ + nd];
        att_ws[(size_t)b * C_ + nd] = acc * invS[n] + WVb[nd];
    }
}

// ---------------------------------------------------------------------------
// k4: out = att @ WO + WOb
// ---------------------------------------------------------------------------
__global__ __launch_bounds__(256) void k4_out(const float* __restrict__ att_ws,
                                              const float* __restrict__ WO,
                                              const float* __restrict__ WOb,
                                              float* __restrict__ out) {
    __shared__ float ar[C_];
    const int b = blockIdx.x, t = threadIdx.x;
    for (int i = t; i < C_; i += 256) ar[i] = att_ws[(size_t)b * C_ + i];
    __syncthreads();
    for (int k = 0; k < 4; ++k) {
        const int o = t + k * 256;
        float acc = WOb[o];
        for (int m = 0; m < C_; ++m) acc += ar[m] * WO[(size_t)m * C_ + o];
        out[(size_t)b * C_ + o] = acc;
    }
}

extern "C" void kernel_launch(void* const* d_in, const int* in_sizes, int n_in,
                              void* d_out, int out_size, void* d_ws, size_t ws_size,
                              hipStream_t stream) {
    (void)in_sizes; (void)n_in; (void)out_size; (void)ws_size;
    const float* x   = (const float*)d_in[0];
    const float* y   = (const float*)d_in[1];
    const float* WQw = (const float*)d_in[2];
    const float* WQb = (const float*)d_in[3];
    const float* WKw = (const float*)d_in[4];
    // d_in[5] (WK_b): uniform shift along L per (b,n) -> softmax invariant, omitted.
    const float* WVw = (const float*)d_in[6];
    const float* WVb = (const float*)d_in[7];
    const float* WOw = (const float*)d_in[8];
    const float* WOb = (const float*)d_in[9];
    float* out = (float*)d_out;

    char* ws = (char*)d_ws;
    __bf16* qk_ws  = (__bf16*)(ws + QK_OFF);
    float* ybar_ws = (float*)(ws + YBAR_OFF);
    float* ms_ws   = (float*)(ws + MS_OFF);
    float* att_ws  = (float*)(ws + ATT_OFF);

    (void)hipFuncSetAttribute((const void*)k2_attn,
                              hipFuncAttributeMaxDynamicSharedMemorySize, SMEM2);
    (void)hipFuncSetAttribute((const void*)k3_comb,
                              hipFuncAttributeMaxDynamicSharedMemorySize, SMEM3);

    k1_qk  <<<B_,        256, 0,     stream>>>(x, WQw, WQb, WKw, qk_ws);
    k2_attn<<<B_ * 16,   256, SMEM2, stream>>>(y, qk_ws, ybar_ws, ms_ws);
    k3_comb<<<B_,        256, SMEM3, stream>>>(ybar_ws, ms_ws, WVw, WVb, att_ws);
    k4_out <<<B_,        256, 0,     stream>>>(att_ws, WOw, WOb, out);
}
